// MultiResolutionCodebooks_89850715833208
// MI455X (gfx1250) — matrix operand order repaired and checked by reading, not compiled, not run on hardware
//
#include <hip/hip_runtime.h>

// ---------------------------------------------------------------------------
// Types for CDNA5 WMMA (wave32): V_WMMA_F32_16X16X32_BF16
// ---------------------------------------------------------------------------
typedef __attribute__((ext_vector_type(16))) __bf16 v16bf;
typedef __attribute__((ext_vector_type(8)))  float  v8f;

#define B_ROWS   65536
#define DIM      512
#define NLEV     4
#define MT       2          // two 16-row M tiles per wave -> 32 rows/wave
#define KSTEPS   (DIM / 32) // 16 k-steps of 32 per 16x16x32 WMMA

// bf16 helpers (round-to-nearest-even), hi/lo split for bf16x3 fp32 emulation
static __device__ __forceinline__ unsigned short f32_to_bf16_rne(float x) {
    unsigned int u = __float_as_uint(x);
    unsigned int r = u + 0x7FFFu + ((u >> 16) & 1u);
    return (unsigned short)(r >> 16);
}
static __device__ __forceinline__ float bf16_bits_to_f32(unsigned short h) {
    return __uint_as_float(((unsigned int)h) << 16);
}

// ---------------------------------------------------------------------------
// Kernel 1: split codebooks into bf16 hi/lo + exact fp32 row norms ||c||^2
// One block per codebook row (3840 rows total across the 4 levels).
// ---------------------------------------------------------------------------
__global__ __launch_bounds__(128) void prep_cb(
    const float* __restrict__ cb0, const float* __restrict__ cb1,
    const float* __restrict__ cb2, const float* __restrict__ cb3,
    unsigned short* __restrict__ cb_hi, unsigned short* __restrict__ cb_lo,
    float* __restrict__ cnorm)
{
    int row = blockIdx.x;             // 0..3839
    const float* src;
    if      (row < 2048) src = cb0 + (long)row * DIM;
    else if (row < 3072) src = cb1 + (long)(row - 2048) * DIM;
    else if (row < 3584) src = cb2 + (long)(row - 3072) * DIM;
    else                 src = cb3 + (long)(row - 3584) * DIM;

    int t = threadIdx.x;
    float ss = 0.f;
    for (int i = t; i < DIM; i += 128) {
        float x = src[i];
        unsigned short hi = f32_to_bf16_rne(x);
        float lof = x - bf16_bits_to_f32(hi);
        cb_hi[(long)row * DIM + i] = hi;
        cb_lo[(long)row * DIM + i] = f32_to_bf16_rne(lof);
        ss += x * x;
    }
    // wave32 reduce + cross-wave via LDS
    for (int off = 16; off > 0; off >>= 1) ss += __shfl_down(ss, off, 32);
    __shared__ float red[4];
    if ((t & 31) == 0) red[t >> 5] = ss;
    __syncthreads();
    if (t == 0) cnorm[row] = red[0] + red[1] + red[2] + red[3];
}

// ---------------------------------------------------------------------------
// Kernel 2: scores = ||c||^2 - 2 * (r @ cb^T) via bf16x3 WMMA, fused argmin.
// 1 wave / block, 32 rows / wave. A-fragments pre-swizzled into LDS in the
// contiguous-K order (k = 16*half + s, as B) so each lane reads one contiguous 32B
// chunk per k-step. B streams from the L2-resident bf16 codebooks in the
// 32x16 B layout (lane -> column, 16 consecutive K per lane).
// 6 independent accumulator chains (3 bf16x3 products x MT) maximize WMMA ILP.
// ---------------------------------------------------------------------------
__global__ __launch_bounds__(32) void rq_argmin(
    const float* __restrict__ r_src, int row_stride,
    const unsigned short* __restrict__ cb_hi,
    const unsigned short* __restrict__ cb_lo,
    const float* __restrict__ cnorm,
    int K, int* __restrict__ sids, int level)
{
    // [hi/lo][mt][kstep][lane][16 bf16]  = 2 * 2*16*32*16 * 2B = 64 KB
    __shared__ __align__(32) unsigned short aLDS[2][MT * KSTEPS * 32 * 16];

    const int lane = threadIdx.x;
    const int rbase = blockIdx.x * (MT * 16);
    const int half = lane >> 4;          // 0: lanes 0-15, 1: lanes 16-31
    const int l16  = lane & 15;
    const int koff = half ? 16 : 0;      // A layout: half-wave K split

    // ---- stage A fragments (fp32 -> bf16 hi/lo, pre-swizzled) ----
    for (int mt = 0; mt < MT; ++mt) {
        const float* rp = r_src + (long)(rbase + mt * 16 + l16) * row_stride;
        for (int ks = 0; ks < KSTEPS; ++ks) {
            unsigned short* dh = &aLDS[0][((mt * KSTEPS + ks) * 32 + lane) * 16];
            unsigned short* dl = &aLDS[1][((mt * KSTEPS + ks) * 32 + lane) * 16];
            int k0 = ks * 32 + koff;
            #pragma unroll
            for (int c = 0; c < 2; ++c) {              // chunks K=k0.., K=k0+8..
                const float4 x0 = *(const float4*)(rp + k0 + c * 8);
                const float4 x1 = *(const float4*)(rp + k0 + c * 8 + 4);
                float xs[8] = {x0.x, x0.y, x0.z, x0.w, x1.x, x1.y, x1.z, x1.w};
                #pragma unroll
                for (int j = 0; j < 8; ++j) {
                    unsigned short hi = f32_to_bf16_rne(xs[j]);
                    dh[c * 8 + j] = hi;
                    dl[c * 8 + j] = f32_to_bf16_rne(xs[j] - bf16_bits_to_f32(hi));
                }
            }
        }
    }
    __syncthreads();

    float best[MT][8];
    int   barg[MT][8];
    #pragma unroll
    for (int mt = 0; mt < MT; ++mt)
        #pragma unroll
        for (int i = 0; i < 8; ++i) { best[mt][i] = 3.4e38f; barg[mt][i] = 0; }

    const int ntiles = K >> 4;
    for (int nt = 0; nt < ntiles; ++nt) {
        const int n = nt * 16 + l16;                       // this lane's column
        const unsigned short* bh = cb_hi + (long)n * DIM;  // cb row n = B col n
        const unsigned short* bl = cb_lo + (long)n * DIM;

        // Prefetch next N-tile's codebook rows (global_prefetch_b8) to cover
        // L2 latency on the dominant B stream.
        if (nt + 1 < ntiles) {
            __builtin_prefetch(bh + (long)16 * DIM, 0, 1);
            __builtin_prefetch(bl + (long)16 * DIM, 0, 1);
        }

        // 3 independent accumulator chains per M tile (hi*hi, hi*lo, lo*hi)
        v8f acc0[MT], acc1[MT], acc2[MT];
        #pragma unroll
        for (int mt = 0; mt < MT; ++mt) { acc0[mt] = {}; acc1[mt] = {}; acc2[mt] = {}; }

        #pragma unroll 4
        for (int ks = 0; ks < KSTEPS; ++ks) {
            const int kb = ks * 32 + (half ? 16 : 0);      // B layout K split
            v16bf b_hi = *(const v16bf*)(bh + kb);
            v16bf b_lo = *(const v16bf*)(bl + kb);
            #pragma unroll
            for (int mt = 0; mt < MT; ++mt) {
                v16bf a_hi = *(const v16bf*)&aLDS[0][((mt * KSTEPS + ks) * 32 + lane) * 16];
                v16bf a_lo = *(const v16bf*)&aLDS[1][((mt * KSTEPS + ks) * 32 + lane) * 16];
                acc0[mt] = __builtin_amdgcn_wmma_f32_16x16x32_bf16(
                    false, a_hi, false, b_hi, (short)0, acc0[mt], false, false);
                acc1[mt] = __builtin_amdgcn_wmma_f32_16x16x32_bf16(
                    false, a_hi, false, b_lo, (short)0, acc1[mt], false, false);
                acc2[mt] = __builtin_amdgcn_wmma_f32_16x16x32_bf16(
                    false, a_lo, false, b_hi, (short)0, acc2[mt], false, false);
            }
        }

        const float cn = cnorm[n];
        #pragma unroll
        for (int mt = 0; mt < MT; ++mt)
            #pragma unroll
            for (int i = 0; i < 8; ++i) {
                float g = acc0[mt][i] + acc1[mt][i] + acc2[mt][i];
                float s = cn - 2.0f * g;                   // score (||r||^2 dropped)
                if (s < best[mt][i]) { best[mt][i] = s; barg[mt][i] = n; }
            }
    }

    // ---- argmin reduce across the 16 lanes holding one row's columns ----
    #pragma unroll
    for (int mt = 0; mt < MT; ++mt) {
        #pragma unroll
        for (int i = 0; i < 8; ++i) {
            float b = best[mt][i]; int a = barg[mt][i];
            #pragma unroll
            for (int off = 8; off >= 1; off >>= 1) {       // xor<16 keeps halves apart
                float ob = __shfl_xor(b, off, 32);
                int   oa = __shfl_xor(a, off, 32);
                if (ob < b || (ob == b && oa < a)) { b = ob; a = oa; } // first-min ties
            }
            best[mt][i] = b; barg[mt][i] = a;
        }
        if (l16 == 0) {
            int rofs = rbase + mt * 16 + (half ? 8 : 0);   // C/D layout: vgpr i -> row
            #pragma unroll
            for (int i = 0; i < 8; ++i)
                sids[(long)(rofs + i) * NLEV + level] = barg[mt][i];
        }
    }
}

// ---------------------------------------------------------------------------
// Kernel 3: gather q = cb[idx] (fp32), write q_stack / r_stack / z_q.
// ---------------------------------------------------------------------------
__global__ __launch_bounds__(256) void gather_update(
    const float* __restrict__ r_src, int r_stride,
    const float* __restrict__ cb,
    const int* __restrict__ sids,
    float* __restrict__ z_q, float* __restrict__ r_stack,
    float* __restrict__ q_stack, int level)
{
    long gid = (long)blockIdx.x * blockDim.x + threadIdx.x;  // B*128 threads
    long row = gid >> 7;
    int  d   = (int)(gid & 127) * 4;
    if (row >= B_ROWS) return;

    int idx = sids[row * NLEV + level];
    float4 q = *(const float4*)(cb + (long)idx * DIM + d);
    float4 r = *(const float4*)(r_src + row * (long)r_stride + d);

    *(float4*)(q_stack + (row * NLEV + level) * DIM + d) = q;
    if (level == 0)
        *(float4*)(r_stack + row * NLEV * DIM + d) = r;     // r_stack[:,0] = h
    if (level < NLEV - 1) {
        float4 rn; rn.x = r.x - q.x; rn.y = r.y - q.y; rn.z = r.z - q.z; rn.w = r.w - q.w;
        *(float4*)(r_stack + (row * NLEV + level + 1) * DIM + d) = rn;
    }
    float4 z;
    if (level == 0) z = q;
    else {
        z = *(const float4*)(z_q + row * (long)DIM + d);
        z.x += q.x; z.y += q.y; z.z += q.z; z.w += q.w;
    }
    *(float4*)(z_q + row * (long)DIM + d) = z;
}

// ---------------------------------------------------------------------------
extern "C" void kernel_launch(void* const* d_in, const int* in_sizes, int n_in,
                              void* d_out, int out_size, void* d_ws, size_t ws_size,
                              hipStream_t stream) {
    (void)in_sizes; (void)n_in; (void)out_size; (void)ws_size;
    const float* h   = (const float*)d_in[0];
    const float* cbs[NLEV] = { (const float*)d_in[1], (const float*)d_in[2],
                               (const float*)d_in[3], (const float*)d_in[4] };

    // d_out layout: z_q [B,512] | SIDs [B,4] (int32 bits) | r_stack [B,4,512] | q_stack [B,4,512]
    float* z_q     = (float*)d_out;
    int*   sids    = (int*)(z_q + (long)B_ROWS * DIM);
    float* r_stack = (float*)(sids + (long)B_ROWS * NLEV);
    float* q_stack = r_stack + (long)B_ROWS * NLEV * DIM;

    // workspace: bf16 hi/lo codebooks + fp32 row norms (~8 MB)
    const long KTOT = 2048 + 1024 + 512 + 256;
    unsigned short* cb_hi = (unsigned short*)d_ws;
    unsigned short* cb_lo = cb_hi + KTOT * DIM;
    float*          cnorm = (float*)(cb_lo + KTOT * DIM);

    prep_cb<<<dim3((unsigned)KTOT), dim3(128), 0, stream>>>(
        cbs[0], cbs[1], cbs[2], cbs[3], cb_hi, cb_lo, cnorm);

    const int  Ks[NLEV]     = {2048, 1024, 512, 256};
    const long cb_off[NLEV] = {0, 2048L * DIM, 3072L * DIM, 3584L * DIM};
    const long cn_off[NLEV] = {0, 2048, 3072, 3584};

    for (int l = 0; l < NLEV; ++l) {
        const float* r_src  = (l == 0) ? h : (r_stack + (long)l * DIM);
        int          stride = (l == 0) ? DIM : NLEV * DIM;
        rq_argmin<<<dim3(B_ROWS / (MT * 16)), dim3(32), 0, stream>>>(
            r_src, stride, cb_hi + cb_off[l], cb_lo + cb_off[l],
            cnorm + cn_off[l], Ks[l], sids, l);
        gather_update<<<dim3((B_ROWS * 128) / 256), dim3(256), 0, stream>>>(
            r_src, stride, cbs[l], sids, z_q, r_stack, q_stack, l);
    }
}
